// WorldModel_90512140796008
// MI455X (gfx1250) — compile-verified, matrix operand order checked by reading, not run on hardware
//
#include <hip/hip_runtime.h>
#include <cstdint>
#include <cstddef>

// ---------------------------------------------------------------------------
// Types for CDNA5 WMMA (wave32)
// ---------------------------------------------------------------------------
typedef _Float16 h16;
typedef __attribute__((ext_vector_type(16))) _Float16 v16h;
typedef __attribute__((ext_vector_type(8)))  _Float16 v8h;
typedef __attribute__((ext_vector_type(8)))  float    v8f;

union AV {
    v16h v;
    v8h  h[2];
    _Float16 e[16];
};

// Model dimensions
#define T1N  50
#define BN   16
#define HIDN 1024
#define SDN  1024
#define EDN  1536
#define ADN  6
#define FDN  1024

// Activation codes
#define ACT_NONE 0
#define ACT_ELU  1
#define ACT_TANH 2
#define ACT_SIGM 3

__device__ __forceinline__ float eluf(float x)     { return x > 0.0f ? x : expm1f(x); }
__device__ __forceinline__ float sigmoidf(float x) { return 1.0f / (1.0f + expf(-x)); }
__device__ __forceinline__ float applyAct(float x, int act) {
    if (act == ACT_ELU)  return eluf(x);
    if (act == ACT_TANH) return tanhf(x);
    if (act == ACT_SIGM) return sigmoidf(x);
    return x;
}

// ---------------------------------------------------------------------------
// WMMA B-fragment friendly index for a [K x Npad] f16 weight panel.
// Within a 32-deep K block, lane (= column n) reads 16 contiguous halves:
//   offset = ((k/32)*Npad + n)*32 + g*16 + e   where g,e follow the ISA
//   16-bit A/B fragment layout (lanes 0-15 own K 0-7 & 16-23, lanes 16-31
//   own K 8-15 & 24-31).
// ---------------------------------------------------------------------------
__device__ __forceinline__ size_t bfrag_idx(int k, int n, int Npad) {
    int kl = k & 31;
    int g  = (kl >> 3) & 1;
    int eo = (kl & 16) ? (8 + (kl & 7)) : (kl & 7);
    return ((size_t)(k >> 5) * (size_t)Npad + (size_t)n) * 32 + (size_t)(g * 16 + eo);
}

// ---------------------------------------------------------------------------
// Weight repack kernels (fp32 -> f16 WMMA panel)
// ---------------------------------------------------------------------------
__global__ void repack_lin(const float* __restrict__ W, h16* __restrict__ Bp,
                           int K, int N, int Kpad) {
    long idx = (long)blockIdx.x * blockDim.x + threadIdx.x;
    long tot = (long)Kpad * N;
    if (idx >= tot) return;
    int k = (int)(idx / N), n = (int)(idx % N);
    float v = (k < K) ? W[(size_t)k * N + n] : 0.0f;
    Bp[bfrag_idx(k, n, N)] = (h16)v;
}

// flip==0: conv weight (O,I,kh,kw).  flip==1: deconv weight (I,O,kh,kw), flipped.
__global__ void repack_convw(const float* __restrict__ W, h16* __restrict__ Bp,
                             int Cout, int Cin, int KW, int Npad, int Kpad, int flip) {
    long idx = (long)blockIdx.x * blockDim.x + threadIdx.x;
    long tot = (long)Kpad * Npad;
    if (idx >= tot) return;
    int k = (int)(idx / Npad), n = (int)(idx % Npad);
    int KK = Cin * KW * KW;
    float v = 0.0f;
    if (k < KK && n < Cout) {
        int ci = k / (KW * KW);
        int rr = k - ci * KW * KW;
        int ky = rr / KW, kx = rr - ky * KW;
        if (!flip) v = W[(((size_t)n * Cin + ci) * KW + ky) * KW + kx];
        else       v = W[(((size_t)ci * Cout + n) * KW + (KW - 1 - ky)) * KW + (KW - 1 - kx)];
    }
    Bp[bfrag_idx(k, n, Npad)] = (h16)v;
}

// ---------------------------------------------------------------------------
// Dense GEMM: C[M,N] = act(A[M,K] * B[K,N] + bias)   (A f16 row-major,
// B pre-packed f16 panel, accum fp32 via v_wmma_f32_16x16x32_f16).
// M % 16 == 0, K % 32 == 0, N % 16 == 0.  One 16x16 tile per wave.
// ---------------------------------------------------------------------------
__global__ void gemm_f16_wmma(const h16* __restrict__ A, const h16* __restrict__ B,
                              const float* __restrict__ bias,
                              float* __restrict__ Cf, h16* __restrict__ Ch,
                              int M, int N, int K, int act) {
    const int wave = threadIdx.x >> 5;
    const int lane = threadIdx.x & 31;
    const int tilesN = N >> 4;
    const int tilesM = M >> 4;
    long tile = (long)blockIdx.x * 4 + wave;
    if (tile >= (long)tilesM * tilesN) return;
    const int tm = (int)(tile / tilesN);
    const int tn = (int)(tile % tilesN);
    const int g  = lane >> 4;
    const int mr = lane & 15;
    const int am = tm * 16 + mr;   // row this lane supplies for the A fragment
    const int bn = tn * 16 + mr;   // column this lane supplies for the B fragment
    const h16* Ap = A + (size_t)am * K;
    v8f acc = {};
    const int nkb = K >> 5;
    const h16* b0 = B + (((size_t)0 * N + bn) << 5) + g * 16;
    for (int kb = 0; kb < nkb; ++kb) {
        AV a, b;
        const h16* a0 = Ap + kb * 32 + g * 8;
        a.h[0] = *(const v8h*)(a0);
        a.h[1] = *(const v8h*)(a0 + 16);
        b.h[0] = *(const v8h*)(b0);
        b.h[1] = *(const v8h*)(b0 + 8);
        if (kb + 1 < nkb)
            __builtin_prefetch((const void*)(b0 + (size_t)N * 32), 0, 1);
        acc = __builtin_amdgcn_wmma_f32_16x16x32_f16(false, a.v, false, b.v,
                                                     (short)0, acc, false, false);
        b0 += (size_t)N * 32;
    }
    const float bv = bias ? bias[bn] : 0.0f;
#pragma unroll
    for (int vi = 0; vi < 8; ++vi) {
        int row = tm * 16 + vi + g * 8;   // C layout: VGPR vi -> M = vi (+8 for hi lanes)
        float x = applyAct(acc[vi] + bv, act);
        size_t o = (size_t)row * N + bn;
        Cf[o] = x;
        if (Ch) Ch[o] = (h16)x;
    }
}

// ---------------------------------------------------------------------------
// Implicit-GEMM conv (stride 2, VALID) / deconv (lhs_dilation 2, pad k-1).
// A fragment gathered on the fly from fp32 NCHW input; B pre-packed f16.
// ---------------------------------------------------------------------------
__global__ void conv_gemm_wmma(const float* __restrict__ X, const h16* __restrict__ B,
                               const float* __restrict__ bias, float* __restrict__ Y,
                               int Nimg, int Cin, int IH, int IW,
                               int Cout, int Npad, int KW, int OH, int OW,
                               int Kpad, int act, int isDeconv) {
    const int wave = threadIdx.x >> 5;
    const int lane = threadIdx.x & 31;
    const int M = Nimg * OH * OW;
    const int tilesN = Npad >> 4;
    const long tilesM = (long)((M + 15) >> 4);
    long tile = (long)blockIdx.x * 4 + wave;
    if (tile >= tilesM * tilesN) return;
    const int tm = (int)(tile / tilesN);
    const int tn = (int)(tile % tilesN);
    const int g  = lane >> 4;
    const int mr = lane & 15;
    const int am = tm * 16 + mr;
    const bool arow_ok = (am < M);
    int img = 0, oy = 0, ox = 0;
    if (arow_ok) {
        img = am / (OH * OW);
        int r = am - img * OH * OW;
        oy = r / OW; ox = r - oy * OW;
    }
    const int bn = tn * 16 + mr;
    const int KK = Cin * KW * KW;
    v8f acc = {};
    const int nkb = Kpad >> 5;
    for (int kb = 0; kb < nkb; ++kb) {
        AV a, b;
        int kbase = kb * 32;
#pragma unroll
        for (int e = 0; e < 16; ++e) {
            int klocal = (e < 8) ? (g * 8 + e) : (16 + g * 8 + (e - 8));
            int k = kbase + klocal;
            float val = 0.0f;
            if (arow_ok && k < KK) {
                int ci = k / (KW * KW);
                int rr = k - ci * KW * KW;
                int ky = rr / KW, kx = rr - ky * KW;
                if (!isDeconv) {
                    int iy = oy * 2 + ky, ix = ox * 2 + kx;
                    val = X[(((size_t)img * Cin + ci) * IH + iy) * IW + ix];
                } else {
                    int py = oy + ky - (KW - 1), px = ox + kx - (KW - 1);
                    if (py >= 0 && px >= 0 && !(py & 1) && !(px & 1)) {
                        int iy = py >> 1, ix = px >> 1;
                        if (iy < IH && ix < IW)
                            val = X[(((size_t)img * Cin + ci) * IH + iy) * IW + ix];
                    }
                }
            }
            a.e[e] = (h16)val;
        }
        const h16* b0 = B + (((size_t)kb * Npad + bn) << 5) + g * 16;
        b.h[0] = *(const v8h*)(b0);
        b.h[1] = *(const v8h*)(b0 + 8);
        acc = __builtin_amdgcn_wmma_f32_16x16x32_f16(false, a.v, false, b.v,
                                                     (short)0, acc, false, false);
    }
    if (bn >= Cout) return;
    const float bv = bias ? bias[bn] : 0.0f;
#pragma unroll
    for (int vi = 0; vi < 8; ++vi) {
        int row = tm * 16 + vi + g * 8;
        if (row >= M) continue;
        int im2 = row / (OH * OW);
        int r2  = row - im2 * OH * OW;
        int oy2 = r2 / OW, ox2 = r2 - oy2 * OW;
        float x = applyAct(acc[vi] + bv, act);
        Y[(((size_t)im2 * Cout + bn) * OH + oy2) * OW + ox2] = x;
    }
}

// ---------------------------------------------------------------------------
// Row LayerNorm (+ optional ELU), one block per row.
// ---------------------------------------------------------------------------
__global__ void ln_act_kernel(const float* __restrict__ X, h16* __restrict__ Yh,
                              float* __restrict__ Yf, int N, float eps, int act) {
    const int row = blockIdx.x;
    const float* x = X + (size_t)row * N;
    __shared__ float sbuf[256], s2buf[256];
    float s = 0.0f, s2 = 0.0f;
    for (int i = threadIdx.x; i < N; i += blockDim.x) {
        float v = x[i]; s += v; s2 += v * v;
    }
    sbuf[threadIdx.x] = s; s2buf[threadIdx.x] = s2;
    __syncthreads();
    for (int st = 128; st > 0; st >>= 1) {
        if ((int)threadIdx.x < st) {
            sbuf[threadIdx.x]  += sbuf[threadIdx.x + st];
            s2buf[threadIdx.x] += s2buf[threadIdx.x + st];
        }
        __syncthreads();
    }
    float mean = sbuf[0] / N;
    float var  = s2buf[0] / N - mean * mean;
    float inv  = rsqrtf(var + eps);
    for (int i = threadIdx.x; i < N; i += blockDim.x) {
        float v = (x[i] - mean) * inv;
        if (act == ACT_ELU) v = eluf(v);
        if (Yh) Yh[(size_t)row * N + i] = (h16)v;
        if (Yf) Yf[(size_t)row * N + i] = v;
    }
}

// ---------------------------------------------------------------------------
// GRU gate fusion (reads ln(g) [16,3072]), updates hidden, scatters outputs.
// ---------------------------------------------------------------------------
__global__ void gru_gates(const float* __restrict__ g, float* __restrict__ hidden,
                          h16* __restrict__ hidden_h, float* __restrict__ out_hid,
                          float* __restrict__ out_states, h16* __restrict__ states_h) {
    int idx = blockIdx.x * blockDim.x + threadIdx.x;
    if (idx >= BN * HIDN) return;
    int b = idx >> 10, j = idx & 1023;
    const float* gr = g + (size_t)b * 3 * HIDN;
    float reset = sigmoidf(gr[j]);
    float cand  = eluf(reset * gr[HIDN + j]);
    float upd   = sigmoidf(gr[2 * HIDN + j] - 1.0f);
    float h = upd * cand + (1.0f - upd) * hidden[idx];
    hidden[idx]   = h;
    hidden_h[idx] = (h16)h;
    out_hid[idx]  = h;
    out_states[(size_t)b * 2048 + 1024 + j] = h;
    states_h[(size_t)b * 2048 + 1024 + j]   = (h16)h;
}

// ---------------------------------------------------------------------------
// Straight-through categorical sample: forward value is exactly one-hot.
// Deterministic hash-based uniform (reference PRNG not reproducible).
// ---------------------------------------------------------------------------
__device__ __forceinline__ float hash_u01(uint32_t x) {
    x ^= x >> 17; x *= 0xed5ad4bbu; x ^= x >> 11;
    x *= 0xac4c1b51u; x ^= x >> 15; x *= 0x31848babu; x ^= x >> 14;
    return (float)(x >> 8) * (1.0f / 16777216.0f);
}

__global__ void sample_z(const float* __restrict__ post, h16* __restrict__ z_h,
                         float* __restrict__ out_z, float* __restrict__ out_states,
                         h16* __restrict__ states_h, int t) {
    int idx = blockIdx.x * blockDim.x + threadIdx.x;
    if (idx >= BN * 32) return;
    int b = idx >> 5, c = idx & 31;
    const float* lg = post + (size_t)b * 1024 + c * 32;
    float mx = lg[0];
    for (int i = 1; i < 32; ++i) mx = fmaxf(mx, lg[i]);
    float ex[32], sum = 0.0f;
    for (int i = 0; i < 32; ++i) { ex[i] = expf(lg[i] - mx); sum += ex[i]; }
    float u = hash_u01((uint32_t)(t * 131071 + idx * 2654435761u + 12345u)) * sum;
    int s = 31; float a = 0.0f;
    for (int i = 0; i < 32; ++i) { a += ex[i]; if (u <= a) { s = i; break; } }
    for (int i = 0; i < 32; ++i) {
        float v = (i == s) ? 1.0f : 0.0f;
        int col = c * 32 + i;
        z_h[(size_t)b * 1024 + col]          = (h16)v;
        out_z[(size_t)b * 1024 + col]        = v;
        out_states[(size_t)b * 2048 + col]   = v;
        states_h[(size_t)b * 2048 + col]     = (h16)v;
    }
}

// ---------------------------------------------------------------------------
// Concat builders (f16)
// ---------------------------------------------------------------------------
__global__ void build_az(const float* __restrict__ action, const h16* __restrict__ z_h,
                         h16* __restrict__ az, int t) {
    int idx = blockIdx.x * blockDim.x + threadIdx.x;
    if (idx >= BN * 1056) return;
    int b = idx / 1056, col = idx - b * 1056;
    float v = 0.0f;
    if (col < ADN) {
        if (t > 0) v = action[(((size_t)(t - 1) * BN) + b) * ADN + col];
    } else if (col < ADN + SDN) {
        v = (float)z_h[(size_t)b * SDN + (col - ADN)];
    }
    az[idx] = (h16)v;
}

__global__ void build_ph(const h16* __restrict__ pf_h, const h16* __restrict__ hid_h,
                         h16* __restrict__ ph) {
    int idx = blockIdx.x * blockDim.x + threadIdx.x;
    if (idx >= BN * 2048) return;
    int b = idx >> 11, col = idx & 2047;
    ph[idx] = (col < 1024) ? pf_h[(size_t)b * 1024 + col]
                           : hid_h[(size_t)b * 1024 + (col - 1024)];
}

__global__ void build_he(const h16* __restrict__ hid_h, const h16* __restrict__ emb_h,
                         h16* __restrict__ he, int t) {
    int idx = blockIdx.x * blockDim.x + threadIdx.x;
    if (idx >= BN * 2560) return;
    int b = idx / 2560, col = idx - b * 2560;
    he[idx] = (col < 1024) ? hid_h[(size_t)b * 1024 + col]
                           : emb_h[((size_t)t * BN + b) * EDN + (col - 1024)];
}

__global__ void f32_to_f16(const float* __restrict__ src, h16* __restrict__ dst, long n) {
    long idx = (long)blockIdx.x * blockDim.x + threadIdx.x;
    if (idx < n) dst[idx] = (h16)src[idx];
}

// Final scalar head: out[r] = act(sum_k X[r,k]*w[k] + b)
__global__ void head_kernel(const h16* __restrict__ X, const float* __restrict__ w,
                            const float* __restrict__ b, float* __restrict__ out,
                            int M, int K, int act) {
    int r = blockIdx.x * blockDim.x + threadIdx.x;
    if (r >= M) return;
    float s = b[0];
    const h16* x = X + (size_t)r * K;
    for (int k = 0; k < K; ++k) s += (float)x[k] * w[k];
    out[r] = applyAct(s, act);
}

// ---------------------------------------------------------------------------
// Host orchestration
// ---------------------------------------------------------------------------
static inline long cdivl(long a, long b) { return (a + b - 1) / b; }

extern "C" void kernel_launch(void* const* d_in, const int* in_sizes, int n_in,
                              void* d_out_v, int out_size, void* d_ws, size_t ws_size,
                              hipStream_t stream) {
    (void)in_sizes; (void)n_in; (void)out_size; (void)ws_size;
    // ---- inputs (setup_inputs order) ----
    const float* obs      = (const float*)d_in[0];
    const float* action   = (const float*)d_in[1];
    const float* enc_w[4] = {(const float*)d_in[2], (const float*)d_in[3],
                             (const float*)d_in[4], (const float*)d_in[5]};
    const float* enc_b[4] = {(const float*)d_in[6], (const float*)d_in[7],
                             (const float*)d_in[8], (const float*)d_in[9]};
    const float* rep_w1 = (const float*)d_in[10]; const float* rep_b1 = (const float*)d_in[11];
    const float* rep_w2 = (const float*)d_in[12]; const float* rep_b2 = (const float*)d_in[13];
    const float* pri_w1 = (const float*)d_in[14]; const float* pri_b1 = (const float*)d_in[15];
    const float* pri_w2 = (const float*)d_in[16]; const float* pri_b2 = (const float*)d_in[17];
    const float* tr_w   = (const float*)d_in[18]; const float* tr_b   = (const float*)d_in[19];
    const float* gru_w  = (const float*)d_in[20]; const float* gru_b  = (const float*)d_in[21];
    const float* dec_wpre = (const float*)d_in[22];
    const float* dec_w[4] = {(const float*)d_in[23], (const float*)d_in[24],
                             (const float*)d_in[25], (const float*)d_in[26]};
    const float* dec_b[4] = {(const float*)d_in[27], (const float*)d_in[28],
                             (const float*)d_in[29], (const float*)d_in[30]};
    const float* rew_w[5]; const float* rew_b[5];
    const float* don_w[5]; const float* don_b[5];
    for (int i = 0; i < 5; ++i) {
        rew_w[i] = (const float*)d_in[31 + i];
        rew_b[i] = (const float*)d_in[36 + i];
        don_w[i] = (const float*)d_in[41 + i];
        don_b[i] = (const float*)d_in[46 + i];
    }
    float* out = (float*)d_out_v;

    // ---- output offsets (floats) ----
    const size_t OBS_OFF = 0;
    const size_t REW_OFF = 2457600, DONE_OFF = 2458400;
    const size_t ST_OFF = 2459200, POST_OFF = 4097600, PRI_OFF = 4916800;
    const size_t Z_OFF = 5736000, HID_OFF = 6555200;

    // ---- workspace bump allocator ----
    char* wsb = (char*)d_ws;
    size_t cur = 0;
    auto alloc = [&](size_t bytes) -> void* {
        cur = (cur + 255) & ~(size_t)255;
        void* p = wsb + cur;
        cur += bytes;
        return p;
    };

    const int NB = T1N * BN;  // 800

    // f16 weight panels
    const int encCin[4]  = {3, 48, 96, 192};
    const int encCout[4] = {48, 96, 192, 384};
    const int encKpad[4] = {64, 768, 1536, 3072};
    h16* Wenc[4];
    for (int i = 0; i < 4; ++i)
        Wenc[i] = (h16*)alloc((size_t)encKpad[i] * encCout[i] * sizeof(h16));
    h16* Wtr   = (h16*)alloc((size_t)1056 * 1024 * sizeof(h16));
    h16* Wgru  = (h16*)alloc((size_t)2048 * 3072 * sizeof(h16));
    h16* Wpri1 = (h16*)alloc((size_t)1024 * 1024 * sizeof(h16));
    h16* Wpri2 = (h16*)alloc((size_t)1024 * 1024 * sizeof(h16));
    h16* Wrep1 = (h16*)alloc((size_t)2560 * 1024 * sizeof(h16));
    h16* Wrep2 = (h16*)alloc((size_t)1024 * 1024 * sizeof(h16));
    h16* Wdpre = (h16*)alloc((size_t)2048 * 1536 * sizeof(h16));
    const int decCin[4]  = {1536, 192, 96, 48};
    const int decCout[4] = {192, 96, 48, 3};
    const int decNpad[4] = {192, 96, 48, 16};
    const int decK[4]    = {5, 5, 6, 6};
    const int decKpad[4] = {38400, 4800, 3456, 1728};
    const int decIH[4]   = {1, 5, 13, 30};
    const int decOH[4]   = {5, 13, 30, 64};
    h16* Wdec[4];
    for (int i = 0; i < 4; ++i)
        Wdec[i] = (h16*)alloc((size_t)decKpad[i] * decNpad[i] * sizeof(h16));
    h16* Wrew[4]; h16* Wdon[4];
    Wrew[0] = (h16*)alloc((size_t)2048 * 1024 * sizeof(h16));
    Wdon[0] = (h16*)alloc((size_t)2048 * 1024 * sizeof(h16));
    for (int i = 1; i < 4; ++i) {
        Wrew[i] = (h16*)alloc((size_t)1024 * 1024 * sizeof(h16));
        Wdon[i] = (h16*)alloc((size_t)1024 * 1024 * sizeof(h16));
    }

    // fp32 activations (big buffers aliased across encoder/decoder phases)
    float* convA   = (float*)alloc((size_t)800 * 48 * 31 * 31 * sizeof(float)); // conv1 out / dec3 out
    float* convB   = (float*)alloc((size_t)800 * 96 * 14 * 14 * sizeof(float)); // conv2 out / dec2 out
    float* convC   = (float*)alloc((size_t)800 * 192 * 6 * 6 * sizeof(float));  // conv3 out / dec1 out
    float* embedF  = (float*)alloc((size_t)NB * EDN * sizeof(float));           // conv4 out
    float* dec0F   = (float*)alloc((size_t)NB * 1536 * sizeof(float));
    float* mlpraw  = (float*)alloc((size_t)NB * 1024 * sizeof(float));
    float* hidden  = (float*)alloc((size_t)BN * HIDN * sizeof(float));
    float* pfF     = (float*)alloc((size_t)BN * 1024 * sizeof(float));
    float* grawF   = (float*)alloc((size_t)BN * 3072 * sizeof(float));
    float* glnF    = (float*)alloc((size_t)BN * 3072 * sizeof(float));
    float* t1F     = (float*)alloc((size_t)BN * 1024 * sizeof(float));

    // f16 activations
    h16* embedH  = (h16*)alloc((size_t)NB * EDN * sizeof(h16));
    h16* statesH = (h16*)alloc((size_t)NB * 2048 * sizeof(h16));
    h16* mlpH    = (h16*)alloc((size_t)NB * 1024 * sizeof(h16));
    h16* zH      = (h16*)alloc((size_t)BN * SDN * sizeof(h16));
    h16* hidH    = (h16*)alloc((size_t)BN * HIDN * sizeof(h16));
    h16* pfH     = (h16*)alloc((size_t)BN * 1024 * sizeof(h16));
    h16* h1H     = (h16*)alloc((size_t)BN * 1024 * sizeof(h16));
    h16* azH     = (h16*)alloc((size_t)BN * 1056 * sizeof(h16));
    h16* phH     = (h16*)alloc((size_t)BN * 2048 * sizeof(h16));
    h16* heH     = (h16*)alloc((size_t)BN * 2560 * sizeof(h16));

    // ---- launch helpers ----
    auto rp_lin = [&](const float* W, h16* Bp, int K, int N, int Kpad) {
        long tot = (long)Kpad * N;
        repack_lin<<<dim3((unsigned)cdivl(tot, 256)), 256, 0, stream>>>(W, Bp, K, N, Kpad);
    };
    auto rp_conv = [&](const float* W, h16* Bp, int Cout, int Cin, int KW,
                       int Npad, int Kpad, int flip) {
        long tot = (long)Kpad * Npad;
        repack_convw<<<dim3((unsigned)cdivl(tot, 256)), 256, 0, stream>>>(
            W, Bp, Cout, Cin, KW, Npad, Kpad, flip);
    };
    auto gemm = [&](const h16* A, const h16* B, const float* bias,
                    float* Cf, h16* Ch, int M, int N, int K, int act) {
        long tiles = (long)(M >> 4) * (N >> 4);
        gemm_f16_wmma<<<dim3((unsigned)cdivl(tiles, 4)), 128, 0, stream>>>(
            A, B, bias, Cf, Ch, M, N, K, act);
    };
    auto convg = [&](const float* X, const h16* B, const float* bias, float* Y,
                     int Nimg, int Cin, int IH, int Cout, int Npad, int KW,
                     int OH, int Kpad, int act, int isDeconv) {
        long M = (long)Nimg * OH * OH;
        long tiles = cdivl(M, 16) * (Npad >> 4);
        conv_gemm_wmma<<<dim3((unsigned)cdivl(tiles, 4)), 128, 0, stream>>>(
            X, B, bias, Y, Nimg, Cin, IH, IH, Cout, Npad, KW, OH, OH, Kpad, act, isDeconv);
    };

    // ---- weight repacks ----
    for (int i = 0; i < 4; ++i)
        rp_conv(enc_w[i], Wenc[i], encCout[i], encCin[i], 4, encCout[i], encKpad[i], 0);
    rp_lin(tr_w, Wtr, 1030, 1024, 1056);
    rp_lin(gru_w, Wgru, 2048, 3072, 2048);
    rp_lin(pri_w1, Wpri1, 1024, 1024, 1024);
    rp_lin(pri_w2, Wpri2, 1024, 1024, 1024);
    rp_lin(rep_w1, Wrep1, 2560, 1024, 2560);
    rp_lin(rep_w2, Wrep2, 1024, 1024, 1024);
    rp_lin(dec_wpre, Wdpre, 2048, 1536, 2048);
    for (int i = 0; i < 4; ++i)
        rp_conv(dec_w[i], Wdec[i], decCout[i], decCin[i], decK[i], decNpad[i], decKpad[i], 1);
    rp_lin(rew_w[0], Wrew[0], 2048, 1024, 2048);
    rp_lin(don_w[0], Wdon[0], 2048, 1024, 2048);
    for (int i = 1; i < 4; ++i) {
        rp_lin(rew_w[i], Wrew[i], 1024, 1024, 1024);
        rp_lin(don_w[i], Wdon[i], 1024, 1024, 1024);
    }

    // ---- encoder: obs [800,3,64,64] -> embed [800,1536] ----
    convg(obs,   Wenc[0], enc_b[0], convA,  800,   3, 64,  48,  48, 4, 31, encKpad[0], ACT_ELU, 0);
    convg(convA, Wenc[1], enc_b[1], convB,  800,  48, 31,  96,  96, 4, 14, encKpad[1], ACT_ELU, 0);
    convg(convB, Wenc[2], enc_b[2], convC,  800,  96, 14, 192, 192, 4,  6, encKpad[2], ACT_ELU, 0);
    convg(convC, Wenc[3], enc_b[3], embedF, 800, 192,  6, 384, 384, 4,  2, encKpad[3], ACT_ELU, 0);
    {
        long n = (long)NB * EDN;
        f32_to_f16<<<dim3((unsigned)cdivl(n, 256)), 256, 0, stream>>>(embedF, embedH, n);
    }

    // ---- RSSM init ----
    hipMemsetAsync(hidden, 0, (size_t)BN * HIDN * sizeof(float), stream);
    hipMemsetAsync(hidH,   0, (size_t)BN * HIDN * sizeof(h16),   stream);
    hipMemsetAsync(zH,     0, (size_t)BN * SDN  * sizeof(h16),   stream);

    // ---- RSSM scan ----
    for (int t = 0; t < T1N; ++t) {
        build_az<<<66, 256, 0, stream>>>(action, zH, azH, t);
        gemm(azH, Wtr, tr_b, pfF, pfH, BN, 1024, 1056, ACT_ELU);
        build_ph<<<128, 256, 0, stream>>>(pfH, hidH, phH);
        gemm(phH, Wgru, gru_b, grawF, nullptr, BN, 3072, 2048, ACT_NONE);
        ln_act_kernel<<<BN, 256, 0, stream>>>(grawF, nullptr, glnF, 3072, 1e-5f, ACT_NONE);
        gru_gates<<<64, 256, 0, stream>>>(glnF, hidden, hidH,
                                          out + HID_OFF + (size_t)t * BN * HIDN,
                                          out + ST_OFF + (size_t)t * BN * 2048,
                                          statesH + (size_t)t * BN * 2048);
        // prior
        gemm(hidH, Wpri1, pri_b1, t1F, nullptr, BN, 1024, 1024, ACT_NONE);
        ln_act_kernel<<<BN, 256, 0, stream>>>(t1F, h1H, nullptr, 1024, 1e-5f, ACT_ELU);
        gemm(h1H, Wpri2, pri_b2, out + PRI_OFF + (size_t)t * BN * 1024, nullptr,
             BN, 1024, 1024, ACT_NONE);
        // posterior
        build_he<<<160, 256, 0, stream>>>(hidH, embedH, heH, t);
        gemm(heH, Wrep1, rep_b1, t1F, nullptr, BN, 1024, 2560, ACT_NONE);
        ln_act_kernel<<<BN, 256, 0, stream>>>(t1F, h1H, nullptr, 1024, 1e-5f, ACT_ELU);
        gemm(h1H, Wrep2, rep_b2, out + POST_OFF + (size_t)t * BN * 1024, nullptr,
             BN, 1024, 1024, ACT_NONE);
        // straight-through sample -> z
        sample_z<<<2, 256, 0, stream>>>(out + POST_OFF + (size_t)t * BN * 1024, zH,
                                        out + Z_OFF + (size_t)t * BN * 1024,
                                        out + ST_OFF + (size_t)t * BN * 2048,
                                        statesH + (size_t)t * BN * 2048, t);
    }

    // ---- decoder: states [800,2048] -> obs_mean [800,3,64,64] ----
    gemm(statesH, Wdpre, nullptr, dec0F, nullptr, NB, 1536, 2048, ACT_NONE);
    convg(dec0F, Wdec[0], dec_b[0], convC, 800, decCin[0], decIH[0], decCout[0],
          decNpad[0], decK[0], decOH[0], decKpad[0], ACT_ELU, 1);
    convg(convC, Wdec[1], dec_b[1], convB, 800, decCin[1], decIH[1], decCout[1],
          decNpad[1], decK[1], decOH[1], decKpad[1], ACT_ELU, 1);
    convg(convB, Wdec[2], dec_b[2], convA, 800, decCin[2], decIH[2], decCout[2],
          decNpad[2], decK[2], decOH[2], decKpad[2], ACT_ELU, 1);
    convg(convA, Wdec[3], dec_b[3], out + OBS_OFF, 800, decCin[3], decIH[3], decCout[3],
          decNpad[3], decK[3], decOH[3], decKpad[3], ACT_NONE, 1);

    // ---- reward / done MLP heads ----
    for (int head = 0; head < 2; ++head) {
        h16** W = head ? Wdon : Wrew;
        const float* const* bs = head ? don_b : rew_b;
        const float* wlast = head ? don_w[4] : rew_w[4];
        const float* blast = head ? don_b[4] : rew_b[4];
        float* o = out + (head ? DONE_OFF : REW_OFF);
        int act = head ? ACT_SIGM : ACT_TANH;
        gemm(statesH, W[0], bs[0], mlpraw, nullptr, NB, 1024, 2048, ACT_NONE);
        ln_act_kernel<<<NB, 256, 0, stream>>>(mlpraw, mlpH, nullptr, 1024, 1e-3f, ACT_ELU);
        for (int l = 1; l < 4; ++l) {
            gemm(mlpH, W[l], bs[l], mlpraw, nullptr, NB, 1024, 1024, ACT_NONE);
            ln_act_kernel<<<NB, 256, 0, stream>>>(mlpraw, mlpH, nullptr, 1024, 1e-3f, ACT_ELU);
        }
        head_kernel<<<dim3((unsigned)cdivl(NB, 256)), 256, 0, stream>>>(
            mlpH, wlast, blast, o, NB, 1024, act);
    }
}
